// ModulatedConv2d_13606456394376
// MI455X (gfx1250) — compile-verified, matrix operand order checked by reading, not run on hardware
//
#include <hip/hip_runtime.h>
#include <stdint.h>

// Problem constants (from reference setup_inputs)
#define B_    16
#define CIN   256
#define COUT  256
#define H_    128
#define W_    128
#define S_    512

#define AROWS (W_ + 2)   // 130 LDS rows: guard(-1), m=0..127, guard(128)
#define ASTR  40         // LDS row stride in bf16 elems (80B = 20 banks, conflict-free)

typedef __bf16 v16bf __attribute__((ext_vector_type(16)));
typedef float  v8f   __attribute__((ext_vector_type(8)));
typedef float  v4f   __attribute__((ext_vector_type(4)));

__device__ __forceinline__ unsigned short f32_to_bf16(float f) {
  union { float f; unsigned int u; } c; c.f = f;
  unsigned int r = c.u + 0x7FFFu + ((c.u >> 16) & 1u);  // RNE
  return (unsigned short)(r >> 16);
}
__device__ __forceinline__ unsigned int bf16pk(float a, float b) {
  union { float f; unsigned int u; } ca, cb; ca.f = a; cb.f = b;
  unsigned int ra = (ca.u + 0x7FFFu + ((ca.u >> 16) & 1u)) >> 16;
  unsigned int rb = (cb.u + 0x7FFFu + ((cb.u >> 16) & 1u)) & 0xFFFF0000u;
  return ra | rb;
}

// ---------------------------------------------------------------------------
// Kernel 1: s[b][o] = style[b,:] . fc_w[o,:] + fc_b[o]
// ---------------------------------------------------------------------------
__global__ void style_fc_kernel(const float* __restrict__ style,
                                const float* __restrict__ fc_w,
                                const float* __restrict__ fc_b,
                                float* __restrict__ s) {
  int idx = blockIdx.x * blockDim.x + threadIdx.x;  // 0..4095
  int b = idx >> 8, o = idx & 255;
  const float* sv = style + b * S_;
  const float* wv = fc_w + o * S_;
  float acc = 0.f;
#pragma unroll 8
  for (int k = 0; k < S_; ++k) acc += sv[k] * wv[k];
  s[idx] = acc + fc_b[o];
}

// ---------------------------------------------------------------------------
// Kernel 2: weight transform to bf16, pre-swizzled into WMMA fragment order.
//   wt2 row r = (p*8+cg)*COUT + co holds 32 bf16:
//     elems [0..15]  = lane-half0 K pattern {k0..7, k16..23}
//     elems [16..31] = lane-half1 K pattern {k8..15, k24..31}
//   where k = ci & 31, cg = ci >> 5, p = kh*3+kw.
// ---------------------------------------------------------------------------
__global__ void wt_transform_kernel(const float* __restrict__ conv_w,
                                    unsigned short* __restrict__ wt2) {
  int idx = blockIdx.x * blockDim.x + threadIdx.x;  // 0..589823
  int co = idx & 255;
  int t  = idx >> 8;
  int ci = t & 255;
  int p  = t >> 8;                 // 0..8
  int kh = p / 3, kw = p % 3;
  int cg = ci >> 5, kk = ci & 31;
  int g  = kk >> 3;                // 0..3
  int slot = (((g & 1) << 4) | ((g >> 1) << 3)) + (kk & 7);  // {0,16,8,24}[g]
  float w = conv_w[((co * CIN + ci) * 3 + kh) * 3 + kw];
  long r = (long)(p * 8 + cg) * COUT + co;
  wt2[r * 32 + slot] = f32_to_bf16(w);
}

// ---------------------------------------------------------------------------
// Kernel 3: implicit-GEMM conv, block = 128 pixels (one W row) x all 256 couts.
//   8 waves: waveM 0..3 (32 px), waveN 0..1 (128 couts); 2x8 accumulators/wave.
//   x row loaded to LDS once per (kh, ci-group); 3 kw taps = shifted LDS reads.
//   B fragments read straight from L2-resident pre-swizzled weights.
// ---------------------------------------------------------------------------
__global__ __launch_bounds__(256)
void conv_wmma_kernel(const float* __restrict__ x,
                      const unsigned short* __restrict__ wt2,
                      const float* __restrict__ s,
                      float* __restrict__ out) {
  __shared__ __align__(16) unsigned short As[AROWS * ASTR];

  const int bh  = blockIdx.x;          // 0..2047
  const int b   = bh >> 7;
  const int h   = bh & 127;
  const int tid = threadIdx.x;
  const int wave  = tid >> 5;
  const int lane  = tid & 31;
  const int waveM = wave & 3;          // 32-pixel tile
  const int waveN = wave >> 2;         // 128-cout tile
  const int lrow  = lane & 15;
  const int half  = lane >> 4;

  v8f acc[2][8];
#pragma unroll
  for (int i = 0; i < 2; ++i)
#pragma unroll
    for (int j = 0; j < 8; ++j) {
      v8f z = {0.f, 0.f, 0.f, 0.f, 0.f, 0.f, 0.f, 0.f};
      acc[i][j] = z;
    }

  // zero the two guard rows (m=-1 and m=128) once
  for (int z = tid; z < 2 * ASTR; z += 256) {
    int row = (z < ASTR) ? 0 : (AROWS - 1);
    As[row * ASTR + (z % ASTR)] = 0;
  }

  const int am   = tid & 127;          // pixel index this thread loads
  const int kgrp = tid >> 7;           // 0/1: which 16-channel half

  union Frag { uint4 u[2]; v16bf v; };

  for (int kh = -1; kh <= 1; ++kh) {
    const int y = h + kh;
    if (y < 0 || y >= H_) continue;    // uniform per block: whole plane is zero
    for (int cg = 0; cg < 8; ++cg) {
      const int ci0 = cg * 32;
      // ---- A row -> LDS (pre-swizzled: two aligned 16B runs per thread)
      {
        const float* xp = x + ((long)b * CIN + ci0 + kgrp * 16) * (long)(H_ * W_)
                            + (long)y * W_ + am;
        float f[16];
#pragma unroll
        for (int j = 0; j < 16; ++j) f[j] = xp[j * (H_ * W_)];
        uint4 u0 = make_uint4(bf16pk(f[0], f[1]),  bf16pk(f[2], f[3]),
                              bf16pk(f[4], f[5]),  bf16pk(f[6], f[7]));
        uint4 u1 = make_uint4(bf16pk(f[8], f[9]),  bf16pk(f[10], f[11]),
                              bf16pk(f[12], f[13]), bf16pk(f[14], f[15]));
        char* rowp = (char*)&As[(am + 1) * ASTR];
        *(uint4*)(rowp + kgrp * 16)      = u0;   // slots for k{0..7} / k{16..23}
        *(uint4*)(rowp + 32 + kgrp * 16) = u1;   // slots for k{8..15} / k{24..31}
        if (cg < 7)  // warm next channel group's row
          __builtin_prefetch(xp + 32 * (H_ * W_), 0, 1);
      }
      __syncthreads();

      // ---- 3 kw taps x 8 N-subtiles x 2 M-subtiles = 48 WMMAs
#pragma unroll
      for (int kw = 0; kw < 3; ++kw) {
        Frag a[2];
#pragma unroll
        for (int i = 0; i < 2; ++i) {
          const char* rp = (const char*)&As[(waveM * 32 + i * 16 + lrow + kw) * ASTR];
          a[i].u[0] = *(const uint4*)(rp + half * 32);
          a[i].u[1] = *(const uint4*)(rp + half * 32 + 16);
        }
        const unsigned short* wrow =
            wt2 + (long)((((kh + 1) * 3 + kw) * 8 + cg) * COUT) * 32;
#pragma unroll
        for (int j = 0; j < 8; ++j) {
          const int n = waveN * 128 + j * 16 + lrow;
          Frag bf;
          const char* bp = (const char*)(wrow + (long)n * 32);
          bf.u[0] = *(const uint4*)(bp + half * 32);
          bf.u[1] = *(const uint4*)(bp + half * 32 + 16);
          acc[0][j] = __builtin_amdgcn_wmma_f32_16x16x32_bf16(
              false, a[0].v, false, bf.v, (short)0, acc[0][j], false, false);
          acc[1][j] = __builtin_amdgcn_wmma_f32_16x16x32_bf16(
              false, a[1].v, false, bf.v, (short)0, acc[1][j], false, false);
        }
      }
      __syncthreads();
    }
  }

  // ---- epilogue: out[b][n][h][m] = acc * s[b][n]   (write-once -> nontemporal)
#pragma unroll
  for (int i = 0; i < 2; ++i) {
    const int m0 = waveM * 32 + i * 16 + half * 8;   // 8 contiguous pixels
#pragma unroll
    for (int j = 0; j < 8; ++j) {
      const int n     = waveN * 128 + j * 16 + lrow;
      const float scl = s[b * COUT + n];
      float* op = out + (((long)b * COUT + n) * H_ + h) * (long)W_ + m0;
      v4f lo = {acc[i][j][0] * scl, acc[i][j][1] * scl,
                acc[i][j][2] * scl, acc[i][j][3] * scl};
      v4f hi = {acc[i][j][4] * scl, acc[i][j][5] * scl,
                acc[i][j][6] * scl, acc[i][j][7] * scl};
      __builtin_nontemporal_store(lo, (v4f*)op);
      __builtin_nontemporal_store(hi, (v4f*)(op + 4));
    }
  }
}

// ---------------------------------------------------------------------------
extern "C" void kernel_launch(void* const* d_in, const int* in_sizes, int n_in,
                              void* d_out, int out_size, void* d_ws, size_t ws_size,
                              hipStream_t stream) {
  const float* x      = (const float*)d_in[0];  // (16,256,128,128)
  const float* style  = (const float*)d_in[1];  // (16,512)
  const float* conv_w = (const float*)d_in[2];  // (256,256,3,3)
  const float* fc_w   = (const float*)d_in[3];  // (256,512)
  const float* fc_b   = (const float*)d_in[4];  // (256,)
  float* out = (float*)d_out;                   // (16,256,128,128)

  // ws: [0,16KB) style scales fp32; [16KB, +1.18MB) swizzled bf16 weights
  float*          s_ws  = (float*)d_ws;
  unsigned short* wt_ws = (unsigned short*)((char*)d_ws + 16384);

  style_fc_kernel<<<(B_ * COUT) / 256, 256, 0, stream>>>(style, fc_w, fc_b, s_ws);
  wt_transform_kernel<<<(9 * CIN * COUT) / 256, 256, 0, stream>>>(conv_w, wt_ws);
  conv_wmma_kernel<<<B_ * H_, 256, 0, stream>>>(x, wt_ws, s_ws, out);
}
